// AttentionBlock_bmmtype_19971597926630
// MI455X (gfx1250) — compile-verified, matrix operand order checked by reading, not run on hardware
//
#include <hip/hip_runtime.h>

// Problem constants (match reference)
#define BB  8
#define NN  4096
#define CC  256
#define CIC 128

typedef __attribute__((ext_vector_type(16))) __bf16 v16bf;
typedef __attribute__((ext_vector_type(8)))  float  v8f;
typedef __attribute__((ext_vector_type(4)))  unsigned int u32x4;
typedef __attribute__((ext_vector_type(8)))  int    i32x8;
typedef __attribute__((ext_vector_type(4)))  int    i32x4;
typedef __attribute__((ext_vector_type(4)))  float  f32x4;

union BfPack {
  u32x4 q[2];
  v16bf v;
};

// round-to-nearest-even f32 -> bf16 (raw bits)
__device__ __forceinline__ unsigned short f2bf(float f) {
  unsigned int u = __float_as_uint(f);
  unsigned int r = u + 0x7FFFu + ((u >> 16) & 1u);
  return (unsigned short)(r >> 16);
}
__device__ __forceinline__ unsigned int f2bf2(float lo, float hi) {
  return (unsigned)f2bf(lo) | ((unsigned)f2bf(hi) << 16);
}

// Load a 16x32 bf16 WMMA operand (A or B) from LDS stored row-major-by-index:
// base[idx][k], stride idx_stride (elements). Per ISA 7.12.2 (16-bit A 16x32):
// lane&15 = matrix index (row for A / col for B), lane>>4 selects K-half:
//   elements 0..7  = k in [kk + half*8,      kk + half*8 + 7]
//   elements 8..15 = k in [kk + 16 + half*8, kk + 16 + half*8 + 7]
// -> two aligned 16-byte LDS loads per fragment.
__device__ __forceinline__ v16bf load_frag(const unsigned short* base,
                                           int idx_stride, int idx0, int kk) {
  const int lane = threadIdx.x & 31;
  const int idx  = idx0 + (lane & 15);
  const int lh   = lane >> 4;
  const unsigned short* p = base + idx * idx_stride + kk + lh * 8;
  BfPack pk;
  pk.q[0] = *(const u32x4*)(p);
  pk.q[1] = *(const u32x4*)(p + 16);
  return pk.v;
}

__device__ __forceinline__ v8f wmma_bf16(v16bf a, v16bf b, v8f c) {
  return __builtin_amdgcn_wmma_f32_16x16x32_bf16(false, a, false, b,
                                                 (short)0, c, false, false);
}

// LDS byte offset of a __shared__ pointer: flat addr low 32 bits are the
// workgroup-relative LDS address (ISA 10.2 aperture mapping).
__device__ __forceinline__ unsigned lds_off(const void* p) {
  return (unsigned)(unsigned long long)p;
}

// Issue one TDM 2D tile load global->LDS (bf16 elements, tile == tensor
// window). D# layout per ISA 8.3/8.4: group0 {count=1, lds_addr, global_addr,
// type=2}; group1 {data_size=2B, tensor_dim0/1 == tile_dim0/1, dim0 stride}.
// Tracked by TENSORcnt; EXEC ignored (call from one wave only).
__device__ __forceinline__ void tdm_load_2d(unsigned lds_byte_off,
                                            const void* gptr,
                                            unsigned tile_w_elems,
                                            unsigned tile_h,
                                            unsigned long long row_stride_elems) {
  unsigned long long ga = (unsigned long long)gptr;
  u32x4 g0;
  g0.x = 1u;                                            // count=1, user mode
  g0.y = lds_byte_off;                                  // lds_addr
  g0.z = (unsigned)ga;                                  // global_addr[31:0]
  g0.w = (unsigned)((ga >> 32) & 0x01ffffffu) | (2u << 30);  // ga[56:32], type=2
  i32x8 g1;
  const unsigned td0 = tile_w_elems;                    // tensor_dim0
  const unsigned td1 = tile_h;                          // tensor_dim1
  const unsigned long long s0 = row_stride_elems;       // tensor_dim0_stride
  g1[0] = (int)(1u << 16);                              // data_size=1 -> 2 bytes
  g1[1] = (int)((td0 & 0xffffu) << 16);                 // abar=0 | td0[15:0]
  g1[2] = (int)((td0 >> 16) | ((td1 & 0xffffu) << 16));
  g1[3] = (int)((td1 >> 16) | (tile_w_elems << 16));    // tile_dim0
  g1[4] = (int)(tile_h);                                // tile_dim1, tile_dim2=0
  g1[5] = (int)(unsigned)(s0 & 0xffffffffu);
  g1[6] = (int)(unsigned)(s0 >> 32);                    // + dim1_stride lo16 = 0
  g1[7] = 0;
  i32x4 z4 = (i32x4)0;
#if defined(__clang_major__) && (__clang_major__ >= 23)
  i32x8 z8 = (i32x8)0;
  __builtin_amdgcn_tensor_load_to_lds(g0, g1, z4, z4, z8, 0);
#else
  __builtin_amdgcn_tensor_load_to_lds(g0, g1, z4, z4, 0);
#endif
}

// -------------------- Kernel 1: theta/phi/g projections --------------------
// grid 256 x 256 thr. Block: 128 rows of x resident in LDS (bf16), computes all
// three projections. theta/phi stored [B*N][CI]; g stored TRANSPOSED
// [B*CI][N] (bf16) so the attention kernel can TDM it without a transpose.
__global__ __launch_bounds__(256) void proj_kernel(
    const float* __restrict__ x,
    const float* __restrict__ gw, const float* __restrict__ gb,
    const float* __restrict__ tw, const float* __restrict__ tb,
    const float* __restrict__ pw, const float* __restrict__ pb,
    unsigned short* __restrict__ outT, unsigned short* __restrict__ outP,
    unsigned short* __restrict__ outGT) {
  __shared__ __align__(16) unsigned short Xl[128 * 256];  // 64 KB
  __shared__ __align__(16) unsigned short Wt[128 * 32];   // 8 KB [col][k]

  const int tid  = threadIdx.x;
  const int row0 = blockIdx.x * 128;
  const int wv   = tid >> 5;
  const int lane = tid & 31;
  const int lh   = lane >> 4;
  const int rA   = wv * 16;

  {
    const f32x4* xg = (const f32x4*)(x + (size_t)row0 * CC);
    #pragma unroll
    for (int t = 0; t < 32; ++t) {
      int v = t * 256 + tid;
      f32x4 f = xg[v];
      *(unsigned int*)(&Xl[v * 4])     = f2bf2(f.x, f.y);
      *(unsigned int*)(&Xl[v * 4 + 2]) = f2bf2(f.z, f.w);
    }
  }

  for (int p = 0; p < 3; ++p) {
    const float* W    = (p == 0) ? tw : (p == 1) ? pw : gw;
    const float* bias = (p == 0) ? tb : (p == 1) ? pb : gb;

    v8f acc[8];
    #pragma unroll
    for (int i = 0; i < 8; ++i) acc[i] = (v8f)(0.0f);

    for (int kk = 0; kk < CC; kk += 32) {
      __syncthreads();
      #pragma unroll
      for (int t = 0; t < 16; ++t) {
        int e = t * 256 + tid;
        int r = e >> 7;
        int c = e & 127;
        Wt[c * 32 + r] = f2bf(W[(size_t)(kk + r) * CIC + c]);
      }
      __syncthreads();

      v16bf a = load_frag(Xl, 256, rA, kk);
      #pragma unroll
      for (int ct = 0; ct < 8; ++ct) {
        v16bf b = load_frag(Wt, 32, ct * 16, 0);
        acc[ct] = wmma_bf16(a, b, acc[ct]);
      }
    }

    if (p < 2) {
      unsigned short* O = (p == 0) ? outT : outP;
      #pragma unroll
      for (int ct = 0; ct < 8; ++ct) {
        int col  = ct * 16 + (lane & 15);
        float bv = bias[col];
        #pragma unroll
        for (int r = 0; r < 8; ++r) {
          int m = r + 8 * lh;
          O[(size_t)(row0 + rA + m) * CIC + col] = f2bf(acc[ct][r] + bv);
        }
      }
    } else {
      // g: transposed store. Per lane, acc elements 0..7 are consecutive n,
      // so each tile is one packed 16-byte store.
      const int bb = row0 >> 12;                 // batch
      const int n0 = (row0 & 4095) + rA + 8 * lh;
      #pragma unroll
      for (int ct = 0; ct < 8; ++ct) {
        int ci   = ct * 16 + (lane & 15);
        float bv = bias[ci];
        u32x4 v;
        v.x = f2bf2(acc[ct][0] + bv, acc[ct][1] + bv);
        v.y = f2bf2(acc[ct][2] + bv, acc[ct][3] + bv);
        v.z = f2bf2(acc[ct][4] + bv, acc[ct][5] + bv);
        v.w = f2bf2(acc[ct][6] + bv, acc[ct][7] + bv);
        *(u32x4*)(outGT + ((size_t)(bb * CIC + ci)) * NN + n0) = v;
      }
    }
    __syncthreads();
  }
}

// -------------------- Kernel 2: fused flash attention --------------------
// grid = B*(N/64) blocks x 128 thr (4 waves); wave = 16 queries. K and G^T
// tiles are DMA'd global->LDS by the Tensor Data Mover, double-buffered so the
// next tile's DMA overlaps this tile's 32 WMMAs. Ragged lengths bound the loop.
__global__ __launch_bounds__(128) void attn_kernel(
    const unsigned short* __restrict__ thetaB,
    const unsigned short* __restrict__ phiB,
    const unsigned short* __restrict__ gT,
    const int* __restrict__ lengths, float* __restrict__ y) {
  __shared__ __align__(16) unsigned short Ql[64 * 128];      // 16 KB [q][ci]
  __shared__ __align__(16) unsigned short Kl[2][64 * 128];   // 32 KB [key][ci]
  __shared__ __align__(16) unsigned short Gt[2][128 * 64];   // 32 KB [ci][key]
  __shared__ __align__(16) unsigned short Pl[4][16 * 64];    // 8 KB per-wave P

  const int tid  = threadIdx.x;
  const int b    = blockIdx.x >> 6;
  const int q0   = (blockIdx.x & 63) * 64;
  const int len  = lengths[b];
  const int wv   = tid >> 5;
  const int lane = tid & 31;
  const int lh   = lane >> 4;
  const size_t rowbase = (size_t)b * NN;
  const int kend = (len + 63) & ~63;   // len >= N/2

  if (wv == 0) {
    tdm_load_2d(lds_off(Ql),    thetaB + (rowbase + q0) * CIC, 128, 64, 128);
    tdm_load_2d(lds_off(Kl[0]), phiB + rowbase * CIC,          128, 64, 128);
    tdm_load_2d(lds_off(Gt[0]), gT + (size_t)b * CIC * NN,      64, 128, NN);
  }

  v8f o[8];
  #pragma unroll
  for (int i = 0; i < 8; ++i) o[i] = (v8f)(0.0f);
  float mrow[8], lrow[8];
  #pragma unroll
  for (int r = 0; r < 8; ++r) { mrow[r] = -3.0e38f; lrow[r] = 0.0f; }

  for (int k0 = 0; k0 < kend; k0 += 64) {
    const int ib = (k0 >> 6) & 1;
    if (wv == 0) __builtin_amdgcn_s_wait_tensorcnt(0);
    __syncthreads();
    if (wv == 0 && (k0 + 64) < kend) {   // prefetch next tiles during compute
      tdm_load_2d(lds_off(Kl[ib ^ 1]), phiB + (rowbase + k0 + 64) * CIC,
                  128, 64, 128);
      tdm_load_2d(lds_off(Gt[ib ^ 1]), gT + (size_t)b * CIC * NN + (k0 + 64),
                  64, 128, NN);
    }

    // S = Q K^T (16 q x 64 keys per wave)
    v8f st[4];
    #pragma unroll
    for (int i = 0; i < 4; ++i) st[i] = (v8f)(0.0f);
    #pragma unroll
    for (int c4 = 0; c4 < 4; ++c4) {
      v16bf a = load_frag(Ql, 128, wv * 16, c4 * 32);
      #pragma unroll
      for (int kt = 0; kt < 4; ++kt) {
        v16bf bm = load_frag(Kl[ib], 128, kt * 16, c4 * 32);
        st[kt] = wmma_bf16(a, bm, st[kt]);
      }
    }

    // ragged key mask
    #pragma unroll
    for (int kt = 0; kt < 4; ++kt) {
      int key = k0 + kt * 16 + (lane & 15);
      if (key >= len) {
        #pragma unroll
        for (int r = 0; r < 8; ++r) st[kt][r] = -3.0e38f;
      }
    }

    // online softmax; row stats reduced over the 16-lane halves
    #pragma unroll
    for (int r = 0; r < 8; ++r) {
      float tm = st[0][r];
      #pragma unroll
      for (int kt = 1; kt < 4; ++kt) tm = fmaxf(tm, st[kt][r]);
      tm = fmaxf(tm, __shfl_xor(tm, 1));
      tm = fmaxf(tm, __shfl_xor(tm, 2));
      tm = fmaxf(tm, __shfl_xor(tm, 4));
      tm = fmaxf(tm, __shfl_xor(tm, 8));
      float mnew  = fmaxf(mrow[r], tm);
      float scale = __expf(mrow[r] - mnew);
      float ps = 0.0f;
      #pragma unroll
      for (int kt = 0; kt < 4; ++kt) {
        float pv = __expf(st[kt][r] - mnew);
        st[kt][r] = pv;
        ps += pv;
      }
      ps += __shfl_xor(ps, 1);
      ps += __shfl_xor(ps, 2);
      ps += __shfl_xor(ps, 4);
      ps += __shfl_xor(ps, 8);
      lrow[r] = lrow[r] * scale + ps;
      mrow[r] = mnew;
      #pragma unroll
      for (int ot = 0; ot < 8; ++ot) o[ot][r] *= scale;
    }

    // repack P (C-layout) -> A-fragment layout via per-wave LDS staging
    #pragma unroll
    for (int kt = 0; kt < 4; ++kt) {
      int kl = kt * 16 + (lane & 15);
      #pragma unroll
      for (int r = 0; r < 8; ++r) {
        int m = r + 8 * lh;
        Pl[wv][m * 64 + kl] = f2bf(st[kt][r]);
      }
    }

    // O += P @ G
    #pragma unroll
    for (int kh = 0; kh < 2; ++kh) {
      v16bf a = load_frag(&Pl[wv][0], 64, 0, kh * 32);
      #pragma unroll
      for (int ot = 0; ot < 8; ++ot) {
        v16bf bm = load_frag(Gt[ib], 64, ot * 16, kh * 32);
        o[ot] = wmma_bf16(a, bm, o[ot]);
      }
    }
  }

  #pragma unroll
  for (int r = 0; r < 8; ++r) {
    float inv = 1.0f / lrow[r];
    int m = r + 8 * lh;
    size_t row = rowbase + q0 + wv * 16 + m;
    #pragma unroll
    for (int ot = 0; ot < 8; ++ot)
      y[row * CIC + ot * 16 + (lane & 15)] = o[ot][r] * inv;
  }
}

// ---------- Kernel 3: output projection + bias + residual + mask ----------
__global__ __launch_bounds__(256) void outproj_kernel(
    const float* __restrict__ y,
    const float* __restrict__ Ww, const float* __restrict__ Wb,
    const float* __restrict__ x, const int* __restrict__ lengths,
    float* __restrict__ out) {
  __shared__ __align__(16) unsigned short Yl[64 * 128];    // 16 KB
  __shared__ __align__(16) unsigned short Wt[256 * 128];   // 64 KB [c][k]

  const int tid  = threadIdx.x;
  const int row0 = blockIdx.x * 64;
  const int wv   = tid >> 5;
  const int lane = tid & 31;
  const int lh   = lane >> 4;
  const int rowg = wv & 3;
  const int chl  = wv >> 2;

  #pragma unroll 4
  for (int t = 0; t < 128; ++t) {
    int e = t * 256 + tid;
    int k = e >> 8;
    int c = e & 255;
    Wt[c * 128 + k] = f2bf(Ww[(size_t)k * 256 + c]);
  }
  #pragma unroll
  for (int t = 0; t < 32; ++t) {
    int e = t * 256 + tid;
    int row = e >> 7;
    int col = e & 127;
    Yl[row * 128 + col] = f2bf(y[((size_t)row0 + row) * CIC + col]);
  }
  __syncthreads();

  v8f acc[8];
  #pragma unroll
  for (int i = 0; i < 8; ++i) acc[i] = (v8f)(0.0f);
  #pragma unroll
  for (int kk = 0; kk < CIC; kk += 32) {
    v16bf a = load_frag(Yl, 128, rowg * 16, kk);
    #pragma unroll
    for (int ct = 0; ct < 8; ++ct) {
      v16bf bm = load_frag(Wt, 128, chl * 128 + ct * 16, kk);
      acc[ct] = wmma_bf16(a, bm, acc[ct]);
    }
  }

  #pragma unroll
  for (int ct = 0; ct < 8; ++ct) {
    int col  = chl * 128 + ct * 16 + (lane & 15);
    float bv = Wb[col];
    #pragma unroll
    for (int r = 0; r < 8; ++r) {
      int m = r + 8 * lh;
      size_t row = (size_t)row0 + rowg * 16 + m;
      int bidx = (int)(row >> 12);
      int n    = (int)(row & 4095);
      float v  = acc[ct][r] + bv + x[row * CC + col];
      out[row * CC + col] = (n < lengths[bidx]) ? v : 0.0f;
    }
  }
}

extern "C" void kernel_launch(void* const* d_in, const int* in_sizes, int n_in,
                              void* d_out, int out_size, void* d_ws, size_t ws_size,
                              hipStream_t stream) {
  (void)in_sizes; (void)n_in; (void)out_size; (void)ws_size;
  const float* x       = (const float*)d_in[0];
  const int*   lengths = (const int*)d_in[1];
  const float* g_w     = (const float*)d_in[2];
  const float* g_b     = (const float*)d_in[3];
  const float* theta_w = (const float*)d_in[4];
  const float* theta_b = (const float*)d_in[5];
  const float* phi_w   = (const float*)d_in[6];
  const float* phi_b   = (const float*)d_in[7];
  const float* W_w     = (const float*)d_in[8];
  const float* W_b     = (const float*)d_in[9];
  float* out = (float*)d_out;

  // workspace: theta(bf16) | phi(bf16) | gT(bf16, transposed) | y(f32) ~42 MB
  const size_t projElems = (size_t)BB * NN * CIC;
  unsigned short* thetaB = (unsigned short*)d_ws;
  unsigned short* phiB   = thetaB + projElems;
  unsigned short* gTB    = phiB + projElems;
  float*          yBuf   = (float*)(gTB + projElems);

  proj_kernel<<<(BB * NN) / 128, 256, 0, stream>>>(
      x, g_w, g_b, theta_w, theta_b, phi_w, phi_b, thetaB, phiB, gTB);
  attn_kernel<<<BB * (NN / 64), 128, 0, stream>>>(thetaB, phiB, gTB, lengths, yBuf);
  outproj_kernel<<<(BB * NN) / 64, 256, 0, stream>>>(yBuf, W_w, W_b, x, lengths, out);
}
